// AttentionFlow_11476152615580
// MI455X (gfx1250) — matrix operand order repaired and checked by reading, not compiled, not run on hardware
//
#include <hip/hip_runtime.h>
#include <hip/hip_bf16.h>

// Problem constants (from the reference)
#define BB   16
#define CL   1024
#define QL   128
#define D2   1024
#define OD   4096   // 8d output feature dim

typedef __attribute__((ext_vector_type(16))) _Float16 v16h;
typedef __attribute__((ext_vector_type(8)))  _Float16 v8h;
typedef __attribute__((ext_vector_type(8)))  float    v8f;

// ---------------------------------------------------------------------------
// K0: sq[b*QL + j] = sum_k q[b,j,k] * w_q[k]     (one wave per row)
// ---------------------------------------------------------------------------
__global__ __launch_bounds__(256) void k_sq(const float* __restrict__ q,
                                            const float* __restrict__ wq,
                                            float* __restrict__ ws_sq) {
    const int w = threadIdx.x >> 5, lane = threadIdx.x & 31;
    const int g = blockIdx.x * 8 + w;          // 0 .. B*QL-1
    const int b = g >> 7, j = g & (QL - 1);
    const float* qr = q + ((size_t)b * QL + j) * D2;
    float s = 0.f;
    for (int k = lane; k < D2; k += 32) s += qr[k] * wq[k];
    s += __shfl_xor(s, 16); s += __shfl_xor(s, 8);
    s += __shfl_xor(s, 4);  s += __shfl_xor(s, 2); s += __shfl_xor(s, 1);
    if (lane == 0) ws_sq[g] = s;
}

// ---------------------------------------------------------------------------
// K1: fused  s = (c*w_cq)q^T + sc + sq + bias ; softmax_j ; c2q = a@q
//     writes out slices 0 (c), 1 (c2q), 2 (c*c2q); row-max -> ws_M
//     4 waves / block, 16 context rows per wave, WMMA f32<-f16 16x16x32
// ---------------------------------------------------------------------------
__global__ __launch_bounds__(128) void k_main(const float* __restrict__ c,
                                              const float* __restrict__ q,
                                              const float* __restrict__ wc,
                                              const float* __restrict__ bc,
                                              const float* __restrict__ bq,
                                              const float* __restrict__ wcq,
                                              const float* __restrict__ bcq,
                                              const float* __restrict__ ws_sq,
                                              float* __restrict__ out,
                                              float* __restrict__ ws_M) {
    __shared__ _Float16 qbuf[QL][72];          // 64-wide K slab of q (f16), +8 pad
    __shared__ _Float16 a16[4][16][136];       // per-wave probabilities, +8 pad
    __shared__ float    wcq_sh[D2];
    __shared__ float    wc_sh[D2];
    __shared__ float    sq_sh[QL];
    __shared__ float    sc_sh[64];

    const int tid  = threadIdx.x;
    const int w    = tid >> 5;
    const int lane = tid & 31;
    const int hn   = lane >> 4;                // half-wave select
    const int ln   = lane & 15;
    const int b    = blockIdx.x >> 4;          // 16 blocks per batch
    const int rb   = blockIdx.x & 15;
    const int m0   = rb * 64 + w * 16;         // first context row of this wave

    for (int i = tid; i < D2; i += 128) { wcq_sh[i] = wcq[i]; wc_sh[i] = wc[i]; }
    if (tid < QL) sq_sh[tid] = ws_sq[b * QL + tid];

    const float bias3 = bc[0] + bq[0] + bcq[0];

    const int    mrow = m0 + ln;
    const float* crow = c   + ((size_t)b * CL + mrow) * D2;
    float*       orow = out + ((size_t)b * CL + mrow) * OD;

    v8f acc[8] = {};                           // s tile: 16 rows x 128 q-positions
    float scp = 0.f;                           // partial of c . w_c

    // ---- phase 1: s = (c*w_cq) @ q^T over K = D2 ------------------------------
    for (int kb = 0; kb < D2 / 64; ++kb) {
        __syncthreads();
        // stage q[b][:, kb*64 .. +64) as f16 into LDS (coalesced float4 loads)
        #pragma unroll 4
        for (int ch = 0; ch < 16; ++ch) {
            int flat = (ch * 128 + tid) * 4;   // 0..8191
            int j  = flat >> 6;
            int kl = flat & 63;
            const float4 v = *(const float4*)(q + ((size_t)b * QL + j) * D2 + kb * 64 + kl);
            qbuf[j][kl + 0] = (_Float16)v.x;
            qbuf[j][kl + 1] = (_Float16)v.y;
            qbuf[j][kl + 2] = (_Float16)v.z;
            qbuf[j][kl + 3] = (_Float16)v.w;
        }
        __syncthreads();

        #pragma unroll
        for (int ks = 0; ks < 64; ks += 32) {
            const int k0 = kb * 64 + ks;
            // A fragment (16x32 f16): lane ln = row M, K = 16h + 0..15 contiguous.
            // Fused: copy c -> out slice 0, accumulate sc partial, scale by w_cq.
            v16h af;
            #pragma unroll
            for (int half = 0; half < 2; ++half) {
                const int kk = k0 + 16 * hn + 8 * half;
                float4 x0 = *(const float4*)(crow + kk);
                float4 x1 = *(const float4*)(crow + kk + 4);
                *(float4*)(orow + kk)     = x0;      // out slice 0 = c
                *(float4*)(orow + kk + 4) = x1;
                const float fv[8] = {x0.x, x0.y, x0.z, x0.w, x1.x, x1.y, x1.z, x1.w};
                #pragma unroll
                for (int t = 0; t < 8; ++t) {
                    scp += fv[t] * wc_sh[kk + t];
                    af[half * 8 + t] = (_Float16)(fv[t] * wcq_sh[kk + t]);
                }
            }
            // B fragments (32x16 f16): lane ln = column N=j, K = 16h + 0..15 contiguous
            #pragma unroll
            for (int jt = 0; jt < 8; ++jt) {
                const _Float16* qr = &qbuf[jt * 16 + ln][ks + 16 * hn];
                v8h b0 = *(const v8h*)qr;
                v8h b1 = *(const v8h*)(qr + 8);
                v16h bf;
                #pragma unroll
                for (int t = 0; t < 8; ++t) { bf[t] = b0[t]; bf[8 + t] = b1[t]; }
                acc[jt] = __builtin_amdgcn_wmma_f32_16x16x32_f16(
                    false, af, false, bf, (short)0, acc[jt], false, false);
            }
        }
    }

    // finalize sc: lane L and L+16 each hold half of row (L&15)'s dot
    {
        float scf = scp + __shfl_xor(scp, 16);
        if (lane < 16) sc_sh[w * 16 + ln] = scf;
    }

    // ---- softmax over j (per context row, values across 16 lanes of half-wave)
    #pragma unroll
    for (int r = 0; r < 8; ++r) {
        const float scr = sc_sh[w * 16 + r + 8 * hn];
        float mxv = -1e30f;
        #pragma unroll
        for (int jt = 0; jt < 8; ++jt) {
            float v = acc[jt][r] + scr + sq_sh[jt * 16 + ln] + bias3;
            acc[jt][r] = v;
            mxv = fmaxf(mxv, v);
        }
        mxv = fmaxf(mxv, __shfl_xor(mxv, 1));
        mxv = fmaxf(mxv, __shfl_xor(mxv, 2));
        mxv = fmaxf(mxv, __shfl_xor(mxv, 4));
        mxv = fmaxf(mxv, __shfl_xor(mxv, 8));
        float s = 0.f;
        #pragma unroll
        for (int jt = 0; jt < 8; ++jt) {
            float p = __expf(acc[jt][r] - mxv);
            acc[jt][r] = p;
            s += p;
        }
        s += __shfl_xor(s, 1); s += __shfl_xor(s, 2);
        s += __shfl_xor(s, 4); s += __shfl_xor(s, 8);
        const float inv = 1.f / s;
        #pragma unroll
        for (int jt = 0; jt < 8; ++jt)
            a16[w][r + 8 * hn][jt * 16 + ln] = (_Float16)(acc[jt][r] * inv);
        if (ln == 0) ws_M[b * CL + m0 + r + 8 * hn] = mxv;  // row max for b-softmax
    }

    // ---- phase 2: c2q = a @ q, tiled over 64 output feature columns -----------
    for (int kt = 0; kt < D2 / 64; ++kt) {
        __syncthreads();
        #pragma unroll 4
        for (int ch = 0; ch < 16; ++ch) {
            int flat = (ch * 128 + tid) * 4;
            int j  = flat >> 6;
            int kl = flat & 63;
            const float4 v = *(const float4*)(q + ((size_t)b * QL + j) * D2 + kt * 64 + kl);
            qbuf[j][kl + 0] = (_Float16)v.x;
            qbuf[j][kl + 1] = (_Float16)v.y;
            qbuf[j][kl + 2] = (_Float16)v.z;
            qbuf[j][kl + 3] = (_Float16)v.w;
        }
        __syncthreads();

        v8f acc2[4] = {};
        #pragma unroll
        for (int js = 0; js < 4; ++js) {       // K = j, 4 steps of 32
            // A fragment from probabilities in LDS (row = ln, j = 8h + 0..7, then 16 + 8h + 0..7)
            v8h a0 = *(const v8h*)&a16[w][ln][js * 32 + 8 * hn];
            v8h a1 = *(const v8h*)&a16[w][ln][js * 32 + 16 + 8 * hn];
            v16h af2;
            #pragma unroll
            for (int t = 0; t < 8; ++t) { af2[t] = a0[t]; af2[8 + t] = a1[t]; }
            #pragma unroll
            for (int nt = 0; nt < 4; ++nt) {   // 4 column tiles of 16
                const int col = nt * 16 + ln;
                v16h bf2;
                #pragma unroll
                for (int t = 0; t < 16; ++t)
                    bf2[t] = qbuf[js * 32 + 8 * hn + (t & 7) + 16 * (t >> 3)][col];  // conflict-free gather
                acc2[nt] = __builtin_amdgcn_wmma_f32_16x16x32_f16(
                    false, af2, false, bf2, (short)0, acc2[nt], false, false);
            }
        }
        // epilogue: out slice 1 = c2q, slice 2 = c * c2q
        #pragma unroll
        for (int nt = 0; nt < 4; ++nt) {
            const int col = kt * 64 + nt * 16 + ln;
            #pragma unroll
            for (int r = 0; r < 8; ++r) {
                const int m = m0 + r + 8 * hn;
                const float c2qv = acc2[nt][r];
                const float cv   = c[((size_t)b * CL + m) * D2 + col];
                float* ob = out + ((size_t)b * CL + m) * OD;
                ob[1024 + col] = c2qv;
                ob[2048 + col] = cv * c2qv;
            }
        }
    }
}

// ---------------------------------------------------------------------------
// K2a: per batch, softmax over CL of row-maxes -> ws_w ; zero ws_q2c
// ---------------------------------------------------------------------------
__global__ __launch_bounds__(256) void k_bsoft(const float* __restrict__ ws_M,
                                               float* __restrict__ ws_w,
                                               float* __restrict__ ws_q2c) {
    __shared__ float red[256];
    const int b = blockIdx.x, tid = threadIdx.x;
    const float* Mb = ws_M + (size_t)b * CL;
    float v[4], m = -1e30f;
    #pragma unroll
    for (int i = 0; i < 4; ++i) { v[i] = Mb[tid * 4 + i]; m = fmaxf(m, v[i]); }
    red[tid] = m; __syncthreads();
    for (int s2 = 128; s2 > 0; s2 >>= 1) {
        if (tid < s2) red[tid] = fmaxf(red[tid], red[tid + s2]);
        __syncthreads();
    }
    const float mx = red[0]; __syncthreads();
    float sum = 0.f;
    #pragma unroll
    for (int i = 0; i < 4; ++i) { v[i] = __expf(v[i] - mx); sum += v[i]; }
    red[tid] = sum; __syncthreads();
    for (int s2 = 128; s2 > 0; s2 >>= 1) {
        if (tid < s2) red[tid] += red[tid + s2];
        __syncthreads();
    }
    const float inv = 1.f / red[0];
    #pragma unroll
    for (int i = 0; i < 4; ++i) ws_w[(size_t)b * CL + tid * 4 + i] = v[i] * inv;
    for (int k = tid; k < D2; k += 256) ws_q2c[(size_t)b * D2 + k] = 0.f;
}

// ---------------------------------------------------------------------------
// K2b: q2c[b,k] += sum_i ws_w[b,i] * c[b,i,k]   (128-row partials + f32 atomics)
// ---------------------------------------------------------------------------
__global__ __launch_bounds__(256) void k_q2c(const float* __restrict__ c,
                                             const float* __restrict__ ws_w,
                                             float* __restrict__ ws_q2c) {
    const int b  = blockIdx.x >> 3;
    const int i0 = (blockIdx.x & 7) * 128;
    const int k0 = threadIdx.x * 4;
    const float* cb = c + (size_t)b * CL * D2;
    float4 acc = {0.f, 0.f, 0.f, 0.f};
    for (int i = i0; i < i0 + 128; ++i) {
        const float wv = ws_w[(size_t)b * CL + i];
        const float4 cv = *(const float4*)(cb + (size_t)i * D2 + k0);
        acc.x += wv * cv.x; acc.y += wv * cv.y;
        acc.z += wv * cv.z; acc.w += wv * cv.w;
    }
    float* dst = ws_q2c + (size_t)b * D2 + k0;
    unsafeAtomicAdd(dst + 0, acc.x);
    unsafeAtomicAdd(dst + 1, acc.y);
    unsafeAtomicAdd(dst + 2, acc.z);
    unsafeAtomicAdd(dst + 3, acc.w);
}

// ---------------------------------------------------------------------------
// K3: out slice 3 = c * broadcast(q2c)
// ---------------------------------------------------------------------------
__global__ __launch_bounds__(256) void k_out3(const float* __restrict__ c,
                                              const float* __restrict__ ws_q2c,
                                              float* __restrict__ out) {
    const size_t idx = (size_t)blockIdx.x * 256 + threadIdx.x;   // float4 units
    const int k4 = (int)(idx & 255);
    const size_t tmp = idx >> 8;
    const int i = (int)(tmp & 1023);
    const int b = (int)(tmp >> 10);
    const float4 cv = *(const float4*)(c + ((size_t)b * CL + i) * D2 + k4 * 4);
    const float4 qv = *(const float4*)(ws_q2c + (size_t)b * D2 + k4 * 4);
    float4 o = {cv.x * qv.x, cv.y * qv.y, cv.z * qv.z, cv.w * qv.w};
    *(float4*)(out + ((size_t)b * CL + i) * OD + 3072 + k4 * 4) = o;
}

// ---------------------------------------------------------------------------
extern "C" void kernel_launch(void* const* d_in, const int* in_sizes, int n_in,
                              void* d_out, int out_size, void* d_ws, size_t ws_size,
                              hipStream_t stream) {
    (void)in_sizes; (void)n_in; (void)out_size; (void)ws_size;
    const float* c    = (const float*)d_in[0];
    const float* q    = (const float*)d_in[1];
    const float* w_c  = (const float*)d_in[2];
    const float* b_c  = (const float*)d_in[3];
    const float* w_q  = (const float*)d_in[4];
    const float* b_q  = (const float*)d_in[5];
    const float* w_cq = (const float*)d_in[6];
    const float* b_cq = (const float*)d_in[7];
    float* out = (float*)d_out;

    float* ws     = (float*)d_ws;
    float* ws_sq  = ws;                        // B*QL           = 2048
    float* ws_M   = ws_sq + BB * QL;           // B*CL           = 16384
    float* ws_w   = ws_M  + BB * CL;           // B*CL           = 16384
    float* ws_q2c = ws_w  + BB * CL;           // B*D2           = 16384

    k_sq   <<<BB * QL / 8, 256, 0, stream>>>(q, w_q, ws_sq);
    k_main <<<BB * (CL / 64), 128, 0, stream>>>(c, q, w_c, b_c, b_q, w_cq, b_cq,
                                                ws_sq, out, ws_M);
    k_bsoft<<<BB, 256, 0, stream>>>(ws_M, ws_w, ws_q2c);
    k_q2c  <<<BB * 8, 256, 0, stream>>>(c, ws_w, ws_q2c);
    k_out3 <<<(BB * CL * D2 / 4) / 256, 256, 0, stream>>>(c, ws_q2c, out);
}
